// MultiQueryAttention_8272107012808
// MI455X (gfx1250) — matrix operand order repaired and checked by reading, not compiled, not run on hardware
//
#include <hip/hip_runtime.h>

#define BB   256
#define SS   64
#define DD   4096
#define HH   64
#define DKK  64
#define MM   (BB * SS)   // 16384 rows

typedef __attribute__((ext_vector_type(16))) __bf16 v16bf;
typedef __attribute__((ext_vector_type(8)))  float  v8f;

union FragBF {
    uint4  u[2];
    v16bf  v;
};
union AccF {
    float f[8];
    v8f   v;
};

__device__ __forceinline__ unsigned short f2bf(float f) {
    unsigned int u = __float_as_uint(f);
    u += 0x7FFFu + ((u >> 16) & 1u);     // round-to-nearest-even
    return (unsigned short)(u >> 16);
}
__device__ __forceinline__ float bf2f(unsigned short h) {
    return __uint_as_float(((unsigned int)h) << 16);
}

// ---------------------------------------------------------------------------
// f32 -> bf16 cast, 4 elements / thread, exact-coverage grids (n % 1024 == 0)
// ---------------------------------------------------------------------------
__global__ __launch_bounds__(256)
void cast_f32_to_bf16(const float* __restrict__ src,
                      unsigned short* __restrict__ dst,
                      long long n) {
    long long i = ((long long)blockIdx.x * blockDim.x + threadIdx.x) * 4;
    if (i + 3 < n) {
        float4 f = *(const float4*)(src + i);
        ushort4 o;
        o.x = f2bf(f.x); o.y = f2bf(f.y); o.z = f2bf(f.z); o.w = f2bf(f.w);
        *(ushort4*)(dst + i) = o;
    }
}

// ---------------------------------------------------------------------------
// bf16 WMMA GEMM:  C[m,n] = sum_k A[m,k] * W[n,k] + bias[n]
//   A:  [M,K] bf16 row-major,  W: [N,K] bf16 row-major (so B = W^T, K-major)
//   Each wave computes a 64x64 C tile: 4x4 grid of v_wmma_f32_16x16x32_bf16.
//   A-fragment lane layout (ISA 7.12.2, 16-bit A 16x32):
//     lanes 0-15 : row m0+lane,   K chunks {k0..k0+7, k0+16..k0+23}
//     lanes16-31 : row m0+lane-16,K chunks {k0+8..k0+15, k0+24..k0+31}
//   B-fragment lane layout (32x16, lane = column n):
//     lanes 0-15 : col n0+lane,    K chunks {k0..k0+7, k0+16..k0+23}
//     lanes16-31 : col n0+lane-16, K chunks {k0+8..k0+15, k0+24..k0+31}
// ---------------------------------------------------------------------------
template <bool STORE_BF16>
__global__ __launch_bounds__(256)
void gemm_bf16_wmma(const __bf16* __restrict__ A,
                    const __bf16* __restrict__ W,
                    const float*  __restrict__ bias,
                    void* __restrict__ Cout,
                    int M, int N, int K) {
    const int lane   = threadIdx.x & 31;
    const int lmod   = lane & 15;
    const int lhalf  = lane >> 4;
    const int waveId = blockIdx.x * (blockDim.x >> 5) + (threadIdx.x >> 5);
    const int tilesN = N >> 6;
    const int tm = waveId / tilesN;
    const int tn = waveId - tm * tilesN;
    const int m0 = tm << 6;
    const int n0 = tn << 6;
    if (m0 >= M) return;                 // wave-uniform; EXEC stays all-ones

    AccF acc[4][4];
#pragma unroll
    for (int i = 0; i < 4; ++i)
#pragma unroll
        for (int j = 0; j < 4; ++j)
#pragma unroll
            for (int r = 0; r < 8; ++r) acc[i][j].f[r] = 0.0f;

    const int aKoff = lhalf * 8;    // A first-chunk K offset within 32
    const int bKoff = lhalf * 8;    // B first-chunk K offset within 32

    for (int k0 = 0; k0 < K; k0 += 32) {
        FragBF b[4];
#pragma unroll
        for (int j = 0; j < 4; ++j) {
            const __bf16* bp = W + (size_t)(n0 + j * 16 + lmod) * K + (k0 + bKoff);
            b[j].u[0] = *(const uint4*)(bp);
            b[j].u[1] = *(const uint4*)(bp + 16);
        }
#pragma unroll
        for (int i = 0; i < 4; ++i) {
            FragBF a;
            const __bf16* ap = A + (size_t)(m0 + i * 16 + lmod) * K + (k0 + aKoff);
            a.u[0] = *(const uint4*)(ap);
            a.u[1] = *(const uint4*)(ap + 16);
#pragma unroll
            for (int j = 0; j < 4; ++j) {
                acc[i][j].v = __builtin_amdgcn_wmma_f32_16x16x32_bf16(
                    /*neg_a=*/false, a.v, /*neg_b=*/false, b[j].v,
                    /*c_mod=*/(short)0, acc[i][j].v,
                    /*reuse_a=*/false, /*reuse_b=*/false);
            }
        }
    }

    // Epilogue: C/D layout -> VGPR r holds M = r (lanes 0-15) or 8+r (16-31),
    //           N = n-subtile + lmod.
#pragma unroll
    for (int j = 0; j < 4; ++j) {
        const int col = n0 + j * 16 + lmod;
        const float bv = bias ? bias[col] : 0.0f;
#pragma unroll
        for (int i = 0; i < 4; ++i) {
#pragma unroll
            for (int r = 0; r < 8; ++r) {
                const int row = m0 + i * 16 + lhalf * 8 + r;
                const float val = acc[i][j].f[r] + bv;
                if (STORE_BF16)
                    ((unsigned short*)Cout)[(size_t)row * N + col] = f2bf(val);
                else
                    ((float*)Cout)[(size_t)row * N + col] = val;
            }
        }
    }
}

// ---------------------------------------------------------------------------
// Middle stage (reference is the broadcast-elementwise "attention"):
//   per (b, s=j) row of q2 [M,4096]:
//     thread i in 0..63 handles group l=0..63:
//       sc[l]   = q2[row, i*64+l] * k[b, l, j] / 8
//       w       = softmax_l(sc)
//       mid[row, i*64+l] = w[l] * v[b, j, l]
// ---------------------------------------------------------------------------
__global__ __launch_bounds__(64)
void mqa_middle(const unsigned short* __restrict__ qb,   // [M,4096] bf16
                const float* __restrict__ kf,            // [M,64] f32 (B,S,DK)
                const float* __restrict__ vf,            // [M,64] f32
                unsigned short* __restrict__ midb) {     // [M,4096] bf16
    const int row = blockIdx.x;          // b*64 + s
    const int b   = row >> 6;
    const int s   = row & 63;
    const int t   = threadIdx.x;         // group index i = 0..63

    __shared__ float kcol[64];
    __shared__ float vrow[64];
    kcol[t] = kf[((size_t)(b * 64 + t)) * 64 + s];   // k[b, l=t, dk=s]
    vrow[t] = vf[(size_t)row * 64 + t];              // v[b, s, dk=t]
    __syncthreads();

    const unsigned short* qp = qb + (size_t)row * 4096 + t * 64;
    float sc[64];
    float mx = -3.402823466e38f;
#pragma unroll
    for (int l = 0; l < 64; ++l) {
        float v = bf2f(qp[l]) * kcol[l] * 0.125f;
        sc[l] = v;
        mx = fmaxf(mx, v);
    }
    float sum = 0.0f;
#pragma unroll
    for (int l = 0; l < 64; ++l) {
        float e = __expf(sc[l] - mx);
        sc[l] = e;
        sum += e;
    }
    const float inv = 1.0f / sum;
    unsigned short* op = midb + (size_t)row * 4096 + t * 64;
#pragma unroll
    for (int l = 0; l < 64; ++l)
        op[l] = f2bf(sc[l] * inv * vrow[l]);
}

// ---------------------------------------------------------------------------
extern "C" void kernel_launch(void* const* d_in, const int* in_sizes, int n_in,
                              void* d_out, int out_size, void* d_ws, size_t ws_size,
                              hipStream_t stream) {
    (void)in_sizes; (void)n_in; (void)out_size; (void)ws_size;

    const float* x  = (const float*)d_in[0];
    const float* Wq = (const float*)d_in[1];
    const float* bq = (const float*)d_in[2];
    const float* Wk = (const float*)d_in[3];
    const float* bk = (const float*)d_in[4];
    const float* Wv = (const float*)d_in[5];
    const float* bv = (const float*)d_in[6];
    const float* Wo = (const float*)d_in[7];
    const float* bo = (const float*)d_in[8];

    char* ws = (char*)d_ws;
    size_t off = 0;
    auto wsAlloc = [&](size_t bytes) -> void* {
        void* p = ws + off;
        off += (bytes + 255) & ~(size_t)255;
        return p;
    };

    unsigned short* xb   = (unsigned short*)wsAlloc((size_t)MM  * DD * 2);
    unsigned short* Wqb  = (unsigned short*)wsAlloc((size_t)DD  * DD * 2);
    unsigned short* Wkb  = (unsigned short*)wsAlloc((size_t)DKK * DD * 2);
    unsigned short* Wvb  = (unsigned short*)wsAlloc((size_t)DKK * DD * 2);
    unsigned short* Wob  = (unsigned short*)wsAlloc((size_t)DD  * DD * 2);
    unsigned short* qb   = (unsigned short*)wsAlloc((size_t)MM  * DD * 2);
    float*          kf   = (float*)         wsAlloc((size_t)MM  * DKK * 4);
    float*          vf   = (float*)         wsAlloc((size_t)MM  * DKK * 4);
    unsigned short* midb = (unsigned short*)wsAlloc((size_t)MM  * DD * 2);

    // --- casts (all sizes divisible by 1024) ---
    {
        long long n;
        n = (long long)MM * DD;
        cast_f32_to_bf16<<<(unsigned)(n / 1024), 256, 0, stream>>>(x, xb, n);
        n = (long long)DD * DD;
        cast_f32_to_bf16<<<(unsigned)(n / 1024), 256, 0, stream>>>(Wq, Wqb, n);
        cast_f32_to_bf16<<<(unsigned)(n / 1024), 256, 0, stream>>>(Wo, Wob, n);
        n = (long long)DKK * DD;
        cast_f32_to_bf16<<<(unsigned)(n / 1024), 256, 0, stream>>>(Wk, Wkb, n);
        cast_f32_to_bf16<<<(unsigned)(n / 1024), 256, 0, stream>>>(Wv, Wvb, n);
    }

    // --- q projection: [16384,4096] x [4096,4096]^T -> bf16 q2 ---
    gemm_bf16_wmma<true><<<(MM / 64) * (DD / 64) / 8, 256, 0, stream>>>(
        (const __bf16*)xb, (const __bf16*)Wqb, bq, (void*)qb, MM, DD, DD);

    // --- k / v projections: N = 64, f32 outputs ---
    gemm_bf16_wmma<false><<<(MM / 64) * (DKK / 64) / 8, 256, 0, stream>>>(
        (const __bf16*)xb, (const __bf16*)Wkb, bk, (void*)kf, MM, DKK, DD);
    gemm_bf16_wmma<false><<<(MM / 64) * (DKK / 64) / 8, 256, 0, stream>>>(
        (const __bf16*)xb, (const __bf16*)Wvb, bv, (void*)vf, MM, DKK, DD);

    // --- elementwise scores/softmax/weights*v ---
    mqa_middle<<<MM, 64, 0, stream>>>(qb, kf, vf, midb);

    // --- output projection -> f32 d_out ---
    gemm_bf16_wmma<false><<<(MM / 64) * (DD / 64) / 8, 256, 0, stream>>>(
        (const __bf16*)midb, (const __bf16*)Wob, bo, d_out, MM, DD, DD);
}
